// CategoryConstrainedGRUMN_62577673502781
// MI455X (gfx1250) — matrix operand order repaired and checked by reading, not compiled, not run on hardware
//
#include <hip/hip_runtime.h>
#include <cstdint>
#include <cstddef>

typedef __attribute__((ext_vector_type(16))) _Float16 v16h;
typedef __attribute__((ext_vector_type(8)))  float    v8f;
typedef __attribute__((ext_vector_type(4)))  unsigned int u32x4;
typedef __attribute__((ext_vector_type(4)))  int      i32x4;
typedef __attribute__((ext_vector_type(8)))  int      i32x8;

constexpr int kB  = 32;
constexpr int kS  = 96;
constexpr int kV  = 32000;
constexpr int kE  = 256;
constexpr int kC  = 200;
constexpr int kH  = 512;
constexpr int kA  = 256;
constexpr int kBS = kB * kS;

// ---------------------------------------------------------------------------
// WMMA fragment helpers — wave32 A/B fragments in contiguous k order (k = 16*(lane/16) + s)
// ---------------------------------------------------------------------------
__device__ __forceinline__ v16h frag_A16(const _Float16* __restrict__ base, int ld, int lane)
{
    int m  = lane & 15;
    int hi = (lane >> 4) & 1;
    const _Float16* row = base + (size_t)m * ld;
    v16h a;
#pragma unroll
    for (int j = 0; j < 8; ++j) {
        int k = (hi << 4) + (j << 1);
        a[2 * j]     = row[k];
        a[2 * j + 1] = row[k + 1];
    }
    return a;
}

__device__ __forceinline__ v16h frag_B16(const _Float16* __restrict__ base,
                                         int n_stride, int k_stride, int lane)
{
    int n    = lane & 15;
    int koff = (lane & 16) ? 16 : 0;
    v16h b;
#pragma unroll
    for (int e = 0; e < 16; ++e)
        b[e] = base[(size_t)n * n_stride + (size_t)(koff + e) * k_stride];
    return b;
}

__device__ __forceinline__ v8f wmma16(v16h a, v16h b, v8f c)
{
    return __builtin_amdgcn_wmma_f32_16x16x32_f16(false, a, false, b, (short)0, c, false, false);
}

// Store one 16x16 C tile (f32 and/or f16), optional bias/addend/ReLU.
__device__ __forceinline__ void store_tile(v8f acc, int m0, int n, bool nok, int mo,
                                           const float* __restrict__ bias,
                                           const float* __restrict__ addend, int ldadd,
                                           float* __restrict__ Cf, int ldc,
                                           _Float16* __restrict__ Ch, int ldch, int act)
{
    float bval = (bias && nok) ? bias[n] : 0.f;
#pragma unroll
    for (int r = 0; r < 8; ++r) {
        int m = m0 + mo + r;
        float v = acc[r] + bval;
        if (addend && nok) v += addend[(size_t)m * ldadd + n];
        if (act == 1) v = fmaxf(v, 0.f);
        if (nok) {
            if (Cf) Cf[(size_t)m * ldc + n] = v;
            if (Ch) Ch[(size_t)m * ldch + n] = (_Float16)v;
        }
    }
}

// ---------------------------------------------------------------------------
// Generic WMMA GEMM:  out[M,N] = act( A[M,K](f16) @ W[N,K](f16)^T + bias + addend )
// One 16x64 strip per wave: 4 named accumulators, A-frag reused 4x.
// Tile indices are readfirstlane'd -> scalar branches, EXEC stays all-ones.
// ---------------------------------------------------------------------------
__global__ void wmma_gemm_kernel(const _Float16* __restrict__ A, int lda,
                                 const _Float16* __restrict__ W, int ldw,
                                 const float* __restrict__ bias,
                                 const float* __restrict__ addend, int ldadd,
                                 float* __restrict__ Cf, int ldc,
                                 _Float16* __restrict__ Ch, int ldch,
                                 int M, int N, int K, int act)
{
    int lane   = threadIdx.x & 31;
    int gw     = __builtin_amdgcn_readfirstlane(
                     (int)((blockIdx.x * blockDim.x + threadIdx.x) >> 5));
    int Npad16 = ((N + 15) >> 4) << 4;
    int t64    = (Npad16 + 63) >> 6;
    int tilesM = M >> 4;
    if (gw >= tilesM * t64) return;                 // scalar (wave-uniform) exit
    int tm   = gw / t64, tn64 = gw % t64;
    int nAct = (Npad16 - tn64 * 64) >> 4;           // uniform scalar
    if (nAct > 4) nAct = 4;

    const _Float16* Ab = A + (size_t)(tm * 16) * lda;
    const _Float16* Wb = W + (size_t)(tn64 * 64) * ldw;

    v8f a0 = {}, a1 = {}, a2 = {}, a3 = {};
    if (nAct == 4) {                                // straight-line fast path
        for (int k0 = 0; k0 < K; k0 += 32) {
            __builtin_prefetch(Wb + k0 + 64, 0, 1);
            v16h a  = frag_A16(Ab + k0, lda, lane);
            v16h b0 = frag_B16(Wb + k0, ldw, 1, lane);
            v16h b1 = frag_B16(Wb + (size_t)16 * ldw + k0, ldw, 1, lane);
            v16h b2 = frag_B16(Wb + (size_t)32 * ldw + k0, ldw, 1, lane);
            v16h b3 = frag_B16(Wb + (size_t)48 * ldw + k0, ldw, 1, lane);
            a0 = wmma16(a, b0, a0);
            a1 = wmma16(a, b1, a1);
            a2 = wmma16(a, b2, a2);
            a3 = wmma16(a, b3, a3);
        }
    } else {                                        // uniform-scalar tail path
        for (int k0 = 0; k0 < K; k0 += 32) {
            v16h a = frag_A16(Ab + k0, lda, lane);
            v16h b = frag_B16(Wb + k0, ldw, 1, lane);
            a0 = wmma16(a, b, a0);
            if (nAct > 1) {
                b  = frag_B16(Wb + (size_t)16 * ldw + k0, ldw, 1, lane);
                a1 = wmma16(a, b, a1);
            }
            if (nAct > 2) {
                b  = frag_B16(Wb + (size_t)32 * ldw + k0, ldw, 1, lane);
                a2 = wmma16(a, b, a2);
            }
        }
    }

    int m0 = tm * 16;
    int mo = (lane & 16) ? 8 : 0;
    int nb = tn64 * 64 + (lane & 15);
    store_tile(a0, m0, nb,      nb      < N, mo, bias, addend, ldadd, Cf, ldc, Ch, ldch, act);
    if (nAct > 1) store_tile(a1, m0, nb + 16, nb + 16 < N, mo, bias, addend, ldadd, Cf, ldc, Ch, ldch, act);
    if (nAct > 2) store_tile(a2, m0, nb + 32, nb + 32 < N, mo, bias, addend, ldadd, Cf, ldc, Ch, ldch, act);
    if (nAct > 3) store_tile(a3, m0, nb + 48, nb + 48 < N, mo, bias, addend, ldadd, Cf, ldc, Ch, ldch, act);
}

// ---------------------------------------------------------------------------
// Big-M WMMA GEMM with Tensor-Data-Mover staging of the shared A strip.
// One block = 8 waves = 16x512 output strip sharing a [16 x K] A tile in LDS.
// A tile DMA'd Global->LDS via tensor_load_to_lds (D# per ISA 8.3/8.4),
// waited with s_wait_tensorcnt, consumed via DS reads by all 8 waves.
// Requires lda == K (packed rows), K <= 1024, M multiple of 16.
// ---------------------------------------------------------------------------
__global__ void wmma_gemm_tdm_kernel(const _Float16* __restrict__ A,
                                     const _Float16* __restrict__ W,
                                     const float* __restrict__ bias,
                                     float* __restrict__ Cf, int ldc,
                                     _Float16* __restrict__ Ch, int ldch,
                                     int M, int N, int K, int act, int nbN)
{
    __shared__ _Float16 As[16 * 1024];          // 32 KB, only LDS object -> offset 0
    int lane = threadIdx.x & 31;
    int wave = __builtin_amdgcn_readfirstlane((int)(threadIdx.x >> 5));
    int tm   = blockIdx.x / nbN;
    int nb   = blockIdx.x % nbN;

    // --- Tensor Data Mover: stage A[tm*16 .. tm*16+15][0..K) into LDS ---
    if (threadIdx.x == 0) {
        unsigned long long ga = (unsigned long long)(size_t)(const void*)(A + (size_t)tm * 16 * K);
        unsigned td0 = (unsigned)K;                      // tensor_dim0 (elements)
        unsigned td1 = 16u;                              // tensor_dim1 (rows in strip)
        unsigned long long st0 = (unsigned long long)K;  // dim0 stride (elements)
        u32x4 g0;
        g0[0] = 1u;                                             // count=1, user D#
        g0[1] = 0u;                                             // lds_addr = 0
        g0[2] = (unsigned)ga;                                   // global_addr[31:0]
        g0[3] = (unsigned)((ga >> 32) & 0x01FFFFFFull) | (2u << 30);  // addr[56:32] | type=2
        i32x8 g1;
        g1[0] = (int)(1u << 16);                                // data_size = 2 bytes
        g1[1] = (int)((td0 & 0xFFFFu) << 16);                   // tensor_dim0[15:0]
        g1[2] = (int)(((td0 >> 16) & 0xFFFFu) | ((td1 & 0xFFFFu) << 16));
        g1[3] = (int)(((td1 >> 16) & 0xFFFFu) | ((unsigned)K << 16));  // tile_dim0 = K
        g1[4] = (int)16;                                        // tile_dim1=16, tile_dim2=0
        g1[5] = (int)(unsigned)(st0 & 0xFFFFFFFFull);           // dim0_stride[31:0]
        g1[6] = (int)(unsigned)((st0 >> 32) & 0xFFFFull);       // dim0_stride[47:32]
        g1[7] = 0;
        i32x4 gz = {0, 0, 0, 0};
#if defined(__clang_major__) && (__clang_major__ >= 23)
        i32x8 gz8 = {0, 0, 0, 0, 0, 0, 0, 0};
        __builtin_amdgcn_tensor_load_to_lds(g0, g1, gz, gz, gz8, 0);
#else
        __builtin_amdgcn_tensor_load_to_lds(g0, g1, gz, gz, 0);
#endif
        __builtin_amdgcn_s_wait_tensorcnt(0);
    }
    __syncthreads();

    int Npad16 = ((N + 15) >> 4) << 4;
    int tn64   = nb * 8 + wave;                 // uniform scalar
    if (tn64 * 64 >= Npad16) return;            // after barrier: safe scalar exit
    int nAct = (Npad16 - tn64 * 64) >> 4;
    if (nAct > 4) nAct = 4;

    const _Float16* Wb = W + (size_t)(tn64 * 64) * K;
    v8f a0 = {}, a1 = {}, a2 = {}, a3 = {};
    if (nAct == 4) {
        for (int k0 = 0; k0 < K; k0 += 32) {
            __builtin_prefetch(Wb + k0 + 64, 0, 1);
            v16h a  = frag_A16(As + k0, K, lane);      // DS reads from staged tile
            v16h b0 = frag_B16(Wb + k0, K, 1, lane);
            v16h b1 = frag_B16(Wb + (size_t)16 * K + k0, K, 1, lane);
            v16h b2 = frag_B16(Wb + (size_t)32 * K + k0, K, 1, lane);
            v16h b3 = frag_B16(Wb + (size_t)48 * K + k0, K, 1, lane);
            a0 = wmma16(a, b0, a0);
            a1 = wmma16(a, b1, a1);
            a2 = wmma16(a, b2, a2);
            a3 = wmma16(a, b3, a3);
        }
    } else {
        for (int k0 = 0; k0 < K; k0 += 32) {
            v16h a = frag_A16(As + k0, K, lane);
            v16h b = frag_B16(Wb + k0, K, 1, lane);
            a0 = wmma16(a, b, a0);
            if (nAct > 1) {
                b  = frag_B16(Wb + (size_t)16 * K + k0, K, 1, lane);
                a1 = wmma16(a, b, a1);
            }
            if (nAct > 2) {
                b  = frag_B16(Wb + (size_t)32 * K + k0, K, 1, lane);
                a2 = wmma16(a, b, a2);
            }
        }
    }

    int m0 = tm * 16;
    int mo = (lane & 16) ? 8 : 0;
    int nbase = tn64 * 64 + (lane & 15);
    store_tile(a0, m0, nbase,      nbase      < N, mo, bias, nullptr, 0, Cf, ldc, Ch, ldch, act);
    if (nAct > 1) store_tile(a1, m0, nbase + 16, nbase + 16 < N, mo, bias, nullptr, 0, Cf, ldc, Ch, ldch, act);
    if (nAct > 2) store_tile(a2, m0, nbase + 32, nbase + 32 < N, mo, bias, nullptr, 0, Cf, ldc, Ch, ldch, act);
    if (nAct > 3) store_tile(a3, m0, nbase + 48, nbase + 48 < N, mo, bias, nullptr, 0, Cf, ldc, Ch, ldch, act);
}

// ---------------------------------------------------------------------------
// Utility kernels
// ---------------------------------------------------------------------------
__global__ void cvt_pad_kernel(const float* __restrict__ src, _Float16* __restrict__ dst,
                               int N, int K, int Npad)
{
    size_t i   = (size_t)blockIdx.x * blockDim.x + threadIdx.x;
    size_t tot = (size_t)Npad * K;
    if (i >= tot) return;
    int n = (int)(i / (size_t)K);
    dst[i] = (n < N) ? (_Float16)src[i] : (_Float16)0.f;
}

__global__ void zero_f32_kernel(float* __restrict__ p, size_t n)
{
    size_t i = (size_t)blockIdx.x * blockDim.x + threadIdx.x;
    size_t s = (size_t)gridDim.x * blockDim.x;
    for (; i < n; i += s) p[i] = 0.f;
}

__global__ void zero_f16_kernel(_Float16* __restrict__ p, size_t n)
{
    size_t i = (size_t)blockIdx.x * blockDim.x + threadIdx.x;
    size_t s = (size_t)gridDim.x * blockDim.x;
    for (; i < n; i += s) p[i] = (_Float16)0.f;
}

__global__ void embed_kernel(const int* __restrict__ ids, const int* __restrict__ cids,
                             const float* __restrict__ emb, const float* __restrict__ cemb,
                             _Float16* __restrict__ xH, float* __restrict__ xF,
                             _Float16* __restrict__ ceH)
{
    int tok = blockIdx.x;
    int e   = threadIdx.x;
    int id  = ids[tok];
    int cid = cids[tok];
    float we = emb[(size_t)id * kE + e];
    float cv = cemb[(size_t)cid * kE + e];
    size_t xb = (size_t)tok * (2 * kE);
    xH[xb + e]       = (_Float16)we;
    xH[xb + kE + e]  = (_Float16)cv;
    xF[xb + e]       = we;
    xF[xb + kE + e]  = cv;
    ceH[(size_t)tok * kE + e] = (_Float16)cv;
}

__global__ void copy_ce_kernel(const _Float16* __restrict__ ce, _Float16* __restrict__ ctx)
{
    int tok = blockIdx.x, e = threadIdx.x;
    ctx[(size_t)tok * (kH + kE) + kH + e] = ce[(size_t)tok * kE + e];
}

// ---------------------------------------------------------------------------
// Per-step attention over the memory tape.
// ---------------------------------------------------------------------------
__global__ void att_step_kernel(const _Float16* __restrict__ li,
                                const _Float16* __restrict__ hx_h,
                                const float* __restrict__ hx_f,
                                const _Float16* __restrict__ Wx,
                                const _Float16* __restrict__ Wth,
                                const float* __restrict__ bWx,
                                const float* __restrict__ bWth,
                                const float* __restrict__ memproj,
                                const float* __restrict__ mem,
                                const float* __restrict__ vvec,
                                const float* __restrict__ bv,
                                float* __restrict__ agg_f,
                                _Float16* __restrict__ agg_h,
                                int t)
{
    __shared__ float xh[32][kA];
    __shared__ float sc[32][kS];
    int wave = __builtin_amdgcn_readfirstlane((int)(threadIdx.x >> 5));
    int lane = threadIdx.x & 31;

    // Phase 1: xh = li@Wx^T + hx@Wth^T + biases  (M=32,N=256,K=512)
    for (int tile = wave; tile < 32; tile += 8) {
        int tm = tile >> 4, tn = tile & 15;
        v8f acc = {};
        for (int k0 = 0; k0 < kH; k0 += 32) {
            v16h a = frag_A16(li + (size_t)(tm * 16) * kH + k0, kH, lane);
            v16h b = frag_B16(Wx + (size_t)(tn * 16) * kH + k0, kH, 1, lane);
            acc = wmma16(a, b, acc);
        }
        for (int k0 = 0; k0 < kH; k0 += 32) {
            v16h a = frag_A16(hx_h + (size_t)(tm * 16) * kH + k0, kH, lane);
            v16h b = frag_B16(Wth + (size_t)(tn * 16) * kH + k0, kH, 1, lane);
            acc = wmma16(a, b, acc);
        }
        int n  = tn * 16 + (lane & 15);
        int mo = (lane & 16) ? 8 : 0;
        float bb = bWx[n] + bWth[n];
#pragma unroll
        for (int r = 0; r < 8; ++r) xh[tm * 16 + mo + r][n] = acc[r] + bb;
    }
    __syncthreads();

    // Phase 2: scores for s < t
    float bv0 = bv[0];
    for (int p = threadIdx.x; p < 32 * kS; p += blockDim.x) {
        int b = p / kS, s = p % kS;
        if (s < t) {
            const float* mp = memproj + ((size_t)b * kS + s) * kA;
            float a = bv0;
            for (int j = 0; j < kA; ++j) a += vvec[j] * tanhf(mp[j] + xh[b][j]);
            sc[b][s] = a;
        }
    }
    __syncthreads();

    // Phase 3: softmax per batch row over s<t
    if (threadIdx.x < 32 && t > 0) {
        int b = threadIdx.x;
        float mx = -3.4e38f;
        for (int s = 0; s < t; ++s) mx = fmaxf(mx, sc[b][s]);
        float sum = 0.f;
        for (int s = 0; s < t; ++s) { float e = __expf(sc[b][s] - mx); sc[b][s] = e; sum += e; }
        float inv = 1.f / sum;
        for (int s = 0; s < t; ++s) sc[b][s] *= inv;
    }
    __syncthreads();

    // Phase 4: agg
    for (int p = threadIdx.x; p < 32 * kH; p += blockDim.x) {
        int b = p >> 9, h = p & (kH - 1);
        float a;
        if (t == 0) a = hx_f[(size_t)b * kH + h];
        else {
            a = 0.f;
            const float* mrow = mem + ((size_t)b * kS) * kH + h;
            for (int s = 0; s < t; ++s) a += sc[b][s] * mrow[(size_t)s * kH];
        }
        agg_f[p] = a;
        agg_h[p] = (_Float16)a;
    }
}

// ---------------------------------------------------------------------------
// GRU gates + LayerNorm.
// ---------------------------------------------------------------------------
__global__ void gate_ln_kernel(const float* __restrict__ gi, const float* __restrict__ gh,
                               const float* __restrict__ agg,
                               const float* __restrict__ lng, const float* __restrict__ lnb,
                               float* __restrict__ h_f, _Float16* __restrict__ h_h,
                               float* __restrict__ mem_f, _Float16* __restrict__ seqh_h,
                               int t)
{
    int wave = __builtin_amdgcn_readfirstlane((int)(threadIdx.x >> 5));
    int lane = threadIdx.x & 31;
    for (int b = wave; b < 32; b += 8) {
        const float* gib = gi  + (size_t)b * (3 * kH);
        const float* ghb = gh  + (size_t)b * (3 * kH);
        const float* ab  = agg + (size_t)b * kH;
        float vals[16];
        float s1 = 0.f, s2 = 0.f;
#pragma unroll
        for (int j = 0; j < 16; ++j) {
            int h = lane + 32 * j;
            float r  = 1.f / (1.f + __expf(-(gib[h]        + ghb[h])));
            float z  = 1.f / (1.f + __expf(-(gib[kH + h]   + ghb[kH + h])));
            float nn = tanhf(gib[2 * kH + h] + r * ghb[2 * kH + h]);
            float nh = (1.f - z) * nn + z * ab[h];
            vals[j] = nh; s1 += nh; s2 += nh * nh;
        }
#pragma unroll
        for (int m = 16; m >= 1; m >>= 1) {
            s1 += __shfl_xor(s1, m, 32);
            s2 += __shfl_xor(s2, m, 32);
        }
        float mean = s1 * (1.f / kH);
        float var  = s2 * (1.f / kH) - mean * mean;
        float inv  = rsqrtf(var + 1e-5f);
#pragma unroll
        for (int j = 0; j < 16; ++j) {
            int h = lane + 32 * j;
            float o = (vals[j] - mean) * inv * lng[h] + lnb[h];
            h_f[(size_t)b * kH + h] = o;
            h_h[(size_t)b * kH + h] = (_Float16)o;
            mem_f[((size_t)b * kS + t) * kH + h] = o;
            if (seqh_h) seqh_h[((size_t)b * kS + t) * kH + h] = (_Float16)o;
        }
    }
}

// ---------------------------------------------------------------------------
// Multi-head attention: one block per (batch, head).
// ---------------------------------------------------------------------------
__global__ void mha_kernel(const _Float16* __restrict__ qkv,
                           float* __restrict__ attw,
                           _Float16* __restrict__ aoH)
{
    __shared__ float    sc[kS][kS];
    __shared__ _Float16 sch[kS][kS];
    int b = blockIdx.x >> 3;
    int head = blockIdx.x & 7;
    int wave = __builtin_amdgcn_readfirstlane((int)(threadIdx.x >> 5));
    int lane = threadIdx.x & 31;
    const _Float16* qb = qkv + (size_t)b * kS * (3 * kH) + head * 64;

    // scores = q @ k^T / 8    (M=96, N=96, K=64)
    for (int tile = wave; tile < 36; tile += 8) {
        int tm = tile / 6, tn = tile % 6;
        v8f acc = {};
        for (int k0 = 0; k0 < 64; k0 += 32) {
            v16h a  = frag_A16(qb + (size_t)(tm * 16) * (3 * kH) + k0, 3 * kH, lane);
            v16h bb = frag_B16(qb + kH + (size_t)(tn * 16) * (3 * kH) + k0, 3 * kH, 1, lane);
            acc = wmma16(a, bb, acc);
        }
        int n = tn * 16 + (lane & 15);
        int mo = (lane & 16) ? 8 : 0;
#pragma unroll
        for (int r = 0; r < 8; ++r) sc[tm * 16 + mo + r][n] = acc[r] * 0.125f;
    }
    __syncthreads();

    // row softmax
    for (int i = threadIdx.x; i < kS; i += blockDim.x) {
        float mx = -3.4e38f;
        for (int j = 0; j < kS; ++j) mx = fmaxf(mx, sc[i][j]);
        float sum = 0.f;
        for (int j = 0; j < kS; ++j) { float e = __expf(sc[i][j] - mx); sc[i][j] = e; sum += e; }
        float inv = 1.f / sum;
        for (int j = 0; j < kS; ++j) sc[i][j] *= inv;
    }
    __syncthreads();

    // f16 copy + head-averaged attention weights
    float* aw = attw + (size_t)b * kS * kS;
    for (int p = threadIdx.x; p < kS * kS; p += blockDim.x) {
        float w = sc[p / kS][p % kS];
        sch[p / kS][p % kS] = (_Float16)w;
        atomicAdd(&aw[p], w * 0.125f);
    }
    __syncthreads();

    // ao = attn @ v   (M=96, N=64, K=96)
    for (int tile = wave; tile < 24; tile += 8) {
        int tm = tile / 4, tn = tile % 4;
        v8f acc = {};
        for (int k0 = 0; k0 < kS; k0 += 32) {
            v16h a  = frag_A16(&sch[tm * 16][k0], kS, lane);
            v16h bb = frag_B16(qb + 2 * kH + (size_t)tn * 16 + (size_t)k0 * (3 * kH),
                               1, 3 * kH, lane);
            acc = wmma16(a, bb, acc);
        }
        int n = tn * 16 + (lane & 15);
        int mo = (lane & 16) ? 8 : 0;
#pragma unroll
        for (int r = 0; r < 8; ++r)
            aoH[(size_t)(b * kS + tm * 16 + mo + r) * kH + head * 64 + n] = (_Float16)acc[r];
    }
}

// ---------------------------------------------------------------------------
// Host side
// ---------------------------------------------------------------------------
static inline void launch_gemm(hipStream_t st, const _Float16* A, int lda,
                               const _Float16* W, int ldw,
                               const float* bias, const float* addend, int ldadd,
                               float* Cf, int ldc, _Float16* Ch, int ldch,
                               int M, int N, int K, int act)
{
    int Npad16 = ((N + 15) / 16) * 16;
    int t64    = (Npad16 + 63) / 64;
    int tiles  = (M / 16) * t64;
    int blocks = (tiles + 7) / 8;
    wmma_gemm_kernel<<<blocks, 256, 0, st>>>(A, lda, W, ldw, bias, addend, ldadd,
                                             Cf, ldc, Ch, ldch, M, N, K, act);
}

static inline void launch_gemm_tdm(hipStream_t st, const _Float16* A,
                                   const _Float16* W, const float* bias,
                                   float* Cf, int ldc, _Float16* Ch, int ldch,
                                   int M, int N, int K, int act)
{
    int Npad16 = ((N + 15) / 16) * 16;
    int t64    = (Npad16 + 63) / 64;
    int nbN    = (t64 + 7) / 8;
    wmma_gemm_tdm_kernel<<<(M / 16) * nbN, 256, 0, st>>>(A, W, bias, Cf, ldc, Ch, ldch,
                                                         M, N, K, act, nbN);
}

extern "C" void kernel_launch(void* const* d_in, const int* in_sizes, int n_in,
                              void* d_out, int out_size, void* d_ws, size_t ws_size,
                              hipStream_t stream)
{
    (void)in_sizes; (void)n_in; (void)out_size; (void)ws_size;

    const int*   ids  = (const int*)d_in[0];
    const int*   cids = (const int*)d_in[1];
    const float* emb  = (const float*)d_in[2];
    const float* cemb = (const float*)d_in[3];
    auto F  = [&](int i) { return (const float*)d_in[i]; };
    auto Lp = [&](int l, int o) { return (const float*)d_in[4 + 15 * l + o]; };
    const float* qkv_w = F(34); const float* qkv_b = F(35);
    const float* out_w = F(36); const float* out_b = F(37);
    const float* w1 = F(38);  const float* b1 = F(39);
    const float* w2 = F(40);  const float* b2 = F(41);
    const float* wq = F(42);  const float* bq = F(43);
    const float* wi = F(44);  const float* bi = F(45);
    const float* wc = F(46);  const float* bc = F(47);

    char* ws = (char*)d_ws; size_t off = 0;
    auto alloc = [&](size_t bytes) -> void* {
        void* p = ws + off; off = (off + bytes + 255) & ~(size_t)255; return p;
    };
    auto aH = [&](size_t n) { return (_Float16*)alloc(n * sizeof(_Float16)); };
    auto aF = [&](size_t n) { return (float*)alloc(n * sizeof(float)); };

    _Float16 *projH[2], *wihH[2], *whhH[2], *WhH[2], *WxH[2], *WthH[2];
    for (int l = 0; l < 2; ++l) {
        projH[l] = aH((size_t)512 * 256);
        wihH[l]  = aH((size_t)1536 * 512);
        whhH[l]  = aH((size_t)1536 * 512);
        WhH[l]   = aH((size_t)256 * 512);
        WxH[l]   = aH((size_t)256 * 512);
        WthH[l]  = aH((size_t)256 * 512);
    }
    _Float16* qkvwH = aH((size_t)1536 * 512);
    _Float16* outwH = aH((size_t)512 * 512);
    _Float16* w1H   = aH((size_t)1024 * 768);
    _Float16* w2H   = aH((size_t)512 * 1024);
    _Float16* wqH   = aH((size_t)32000 * 512);
    _Float16* wiH   = aH((size_t)16 * 512);
    _Float16* wcH   = aH((size_t)208 * 512);

    _Float16* xH   = aH((size_t)kBS * 2 * kE);
    float*    xF   = aF((size_t)kBS * 2 * kE);
    _Float16* ceH  = aH((size_t)kBS * kE);
    float* memF[2]; float* mprojF[2]; float* hF[2]; _Float16* hH[2];
    for (int l = 0; l < 2; ++l) {
        memF[l]   = aF((size_t)kB * kS * kH);
        mprojF[l] = aF((size_t)kB * kS * kA);
        hF[l]     = aF((size_t)kB * kH);
        hH[l]     = aH((size_t)kB * kH);
    }
    _Float16* liH  = aH((size_t)kB * kH);
    float*    giF  = aF((size_t)kB * 3 * kH);
    float*    ghF  = aF((size_t)kB * 3 * kH);
    float*    aggF = aF((size_t)kB * kH);
    _Float16* aggH = aH((size_t)kB * kH);
    _Float16* seqhH = aH((size_t)kBS * kH);
    _Float16* qkvH  = aH((size_t)kBS * 3 * kH);
    _Float16* aoH   = aH((size_t)kBS * kH);
    _Float16* ctxH  = aH((size_t)kBS * (kH + kE));
    _Float16* h1H   = aH((size_t)kBS * 1024);
    _Float16* h2H   = aH((size_t)kBS * 512);

    auto cvt = [&](const float* src, _Float16* dst, int N, int K, int Npad) {
        size_t tot = (size_t)Npad * K;
        cvt_pad_kernel<<<(int)((tot + 255) / 256), 256, 0, stream>>>(src, dst, N, K, Npad);
    };
    for (int l = 0; l < 2; ++l) {
        cvt(Lp(l, 0), projH[l], 512, 256, 512);
        cvt(Lp(l, 1), wihH[l], 1536, 512, 1536);
        cvt(Lp(l, 3), whhH[l], 1536, 512, 1536);
        cvt(Lp(l, 5), WhH[l],  256, 512, 256);
        cvt(Lp(l, 7), WxH[l],  256, 512, 256);
        cvt(Lp(l, 9), WthH[l], 256, 512, 256);
    }
    cvt(qkv_w, qkvwH, 1536, 512, 1536);
    cvt(out_w, outwH, 512, 512, 512);
    cvt(w1, w1H, 1024, 768, 1024);
    cvt(w2, w2H, 512, 1024, 512);
    cvt(wq, wqH, 32000, 512, 32000);
    cvt(wi, wiH, 5, 512, 16);
    cvt(wc, wcH, 200, 512, 208);

    embed_kernel<<<kBS, kE, 0, stream>>>(ids, cids, emb, cemb, xH, xF, ceH);

    for (int l = 0; l < 2; ++l) {
        zero_f32_kernel<<<64, 256, 0, stream>>>(hF[l], (size_t)kB * kH);
        zero_f16_kernel<<<64, 256, 0, stream>>>(hH[l], (size_t)kB * kH);
    }
    float* outF   = (float*)d_out;
    size_t OFF_IL = (size_t)kBS * kV;
    size_t OFF_CL = OFF_IL + (size_t)kBS * 5;
    size_t OFF_AW = OFF_CL + (size_t)kBS * kC;
    zero_f32_kernel<<<1152, 256, 0, stream>>>(outF + OFF_AW, (size_t)kB * kS * kS);

    // ---------------- recurrent scan ----------------
    for (int t = 0; t < kS; ++t) {
        for (int l = 0; l < 2; ++l) {
            const _Float16* catA = ceH + (size_t)t * kE;
            const float* add = (l == 0) ? (xF + (size_t)t * 2 * kE) : hF[0];
            int ldadd = (l == 0) ? kS * 2 * kE : kH;
            launch_gemm(stream, catA, kS * kE, projH[l], kE, nullptr, add, ldadd,
                        nullptr, 0, liH, kH, kB, kH, kE, 1);
            launch_gemm(stream, liH, kH, wihH[l], kH, Lp(l, 2), nullptr, 0,
                        giF, 3 * kH, nullptr, 0, kB, 3 * kH, kH, 0);
            att_step_kernel<<<1, 256, 0, stream>>>(liH, hH[l], hF[l], WxH[l], WthH[l],
                                                   Lp(l, 8), Lp(l, 10), mprojF[l], memF[l],
                                                   Lp(l, 11), Lp(l, 12), aggF, aggH, t);
            launch_gemm(stream, aggH, kH, whhH[l], kH, Lp(l, 4), nullptr, 0,
                        ghF, 3 * kH, nullptr, 0, kB, 3 * kH, kH, 0);
            gate_ln_kernel<<<1, 256, 0, stream>>>(giF, ghF, aggF, Lp(l, 13), Lp(l, 14),
                                                  hF[l], hH[l], memF[l],
                                                  (l == 1) ? seqhH : (_Float16*)nullptr, t);
            launch_gemm(stream, hH[l], kH, WhH[l], kH, Lp(l, 6), nullptr, 0,
                        mprojF[l] + (size_t)t * kA, kS * kA, nullptr, 0, kB, kA, kH, 0);
        }
    }

    // ---------------- attention + MLP heads (TDM-staged big GEMMs) ----------
    launch_gemm_tdm(stream, seqhH, qkvwH, qkv_b,
                    nullptr, 0, qkvH, 3 * kH, kBS, 3 * kH, kH, 0);
    mha_kernel<<<kB * 8, 256, 0, stream>>>(qkvH, outF + OFF_AW, aoH);
    launch_gemm_tdm(stream, aoH, outwH, out_b,
                    nullptr, 0, ctxH, kH + kE, kBS, kH, kH, 0);
    copy_ce_kernel<<<kBS, kE, 0, stream>>>(ceH, ctxH);
    launch_gemm_tdm(stream, ctxH, w1H, b1,
                    nullptr, 0, h1H, 1024, kBS, 1024, kH + kE, 1);
    launch_gemm_tdm(stream, h1H, w2H, b2,
                    nullptr, 0, h2H, 512, kBS, 512, 1024, 1);
    launch_gemm_tdm(stream, h2H, wqH, bq,
                    outF, kV, nullptr, 0, kBS, kV, 512, 0);
    launch_gemm_tdm(stream, h2H, wiH, bi,
                    outF + OFF_IL, 5, nullptr, 0, kBS, 5, 512, 0);
    launch_gemm_tdm(stream, h2H, wcH, bc,
                    outF + OFF_CL, kC, nullptr, 0, kBS, kC, 512, 0);
}
